// DensityProximityCrossBlockAttention_22634477650388
// MI455X (gfx1250) — matrix operand order repaired and checked by reading, not compiled, not run on hardware
//
#include <hip/hip_runtime.h>

typedef __attribute__((ext_vector_type(16))) _Float16 v16h;
typedef __attribute__((ext_vector_type(8)))  float    v8f;

namespace {
constexpr int NBLK  = 32;    // patch blocks
constexpr int VTOK  = 98;    // tokens per block
constexpr int LTOK  = 3136;  // total tokens
constexpr int CDIM  = 256;
constexpr int QKVD  = 768;
constexpr int HEADS = 8;
constexpr int HD    = 32;
constexpr int MTILES = LTOK / 16;  // 196
constexpr float SCALE = 0.17677669529663687f;  // 32^-0.5
}

#define WMMA_F16(a, b, c) \
  __builtin_amdgcn_wmma_f32_16x16x32_f16(false, (a), false, (b), (short)0, (c), false, false)

// ---- fragment loaders (contiguous k order: lane half h holds k = 16h+s) -----------------------

__device__ __forceinline__ v16h load_a16(const _Float16* base, int ld, int lane) {
  // A 16x32 f16 fragment from row-major [16 x K] block, row stride ld.
  int m   = lane & 15;
  int kb0 = (lane < 16) ? 0 : 16;
  int kb1 = (lane < 16) ? 8 : 24;
  const _Float16* row = base + m * ld;
  v16h a;
#pragma unroll
  for (int i = 0; i < 8; ++i) a[i] = row[kb0 + i];
#pragma unroll
  for (int i = 0; i < 8; ++i) a[8 + i] = row[kb1 + i];
  return a;
}

__device__ __forceinline__ v16h load_bT16(const _Float16* base, int ld, int lane) {
  // B 32x16 fragment where memory holds B^T row-major: src[n][k], stride ld.
  int n  = lane & 15;
  int kb = (lane < 16) ? 0 : 16;
  const _Float16* row = base + n * ld + kb;
  v16h b;
#pragma unroll
  for (int i = 0; i < 16; ++i) b[i] = row[i];
  return b;
}

__device__ __forceinline__ v16h load_b16(const _Float16* base, int ld, int lane) {
  // B 32x16 fragment from row-major [32 x >=16] source (row = k), stride ld.
  int n  = lane & 15;
  int kb = (lane < 16) ? 0 : 16;
  v16h b;
#pragma unroll
  for (int i = 0; i < 16; ++i) b[i] = base[(kb + i) * ld + n];
  return b;
}

// ---- kernels ---------------------------------------------------------------

__global__ void k_zero(float* dens_raw) {
  if (threadIdx.x < NBLK) dens_raw[threadIdx.x] = 0.0f;
}

__global__ void k_convw(const float* wq, const float* wp, _Float16* wqh, _Float16* wph) {
  int i = blockIdx.x * blockDim.x + threadIdx.x;
  int n1 = QKVD * CDIM, n2 = CDIM * CDIM;
  if (i < n1) wqh[i] = (_Float16)wq[i];
  if (i < n2) wph[i] = (_Float16)wp[i];
}

// rearrange x -> f16 tokens (patch order) + density accumulation
__global__ void k_prep(const float* __restrict__ x, _Float16* __restrict__ tok,
                       float* __restrict__ dens_raw) {
  int l = blockIdx.x;          // token index (n*98 + v)
  int c = threadIdx.x;         // channel
  int n = l / VTOK, v = l % VTOK;
  int t = v / 49, rem = v % 49, hh = rem / 7, ww = rem % 7;
  int nt = n >> 4, nh = (n >> 2) & 3, nw = n & 3;
  int T = nt * 2 + t, H = nh * 7 + hh, W = nw * 7 + ww;
  float val = x[((size_t)(T * 28 + H) * 28 + W) * CDIM + c];
  tok[(size_t)l * CDIM + c] = (_Float16)val;

  float s = val * val;
#pragma unroll
  for (int off = 16; off > 0; off >>= 1) s += __shfl_xor(s, off, 32);
  __shared__ float part[8];
  if ((c & 31) == 0) part[c >> 5] = s;
  __syncthreads();
  if (c == 0) {
    float tot = 0.f;
#pragma unroll
    for (int i = 0; i < 8; ++i) tot += part[i];
    atomicAdd(&dens_raw[n], sqrtf(tot));
  }
}

// density norm + proximity + local window + top-4 -> per-row 32-bit mask
__global__ void k_mask(const float* __restrict__ dens_raw, unsigned* __restrict__ rowmask) {
  int i = threadIdx.x;  // 0..31
  float dm = dens_raw[i] * (1.0f / (float)VTOK);
  float mx = dm;
#pragma unroll
  for (int off = 16; off > 0; off >>= 1) mx = fmaxf(mx, __shfl_xor(mx, off, 32));
  float dn = dm / (mx + 1e-8f);
  __shared__ float dsh[NBLK];
  dsh[i] = dn;
  __syncthreads();

  int nt = i >> 4, nh = (i >> 2) & 3, nw = i & 3;
  float pt = (float)(nt * 2 + 1), ph = (float)(nh * 7 + 3), pw = (float)(nw * 7 + 3);
  float sv[NBLK];
#pragma unroll
  for (int j = 0; j < NBLK; ++j) {
    int jt = j >> 4, jh = (j >> 2) & 3, jw = j & 3;
    float td = fabsf(pt - (float)(jt * 2 + 1)) * 2.0f;   // * pT
    float hd = fabsf(ph - (float)(jh * 7 + 3)) * 7.0f;   // * pH
    float wd = fabsf(pw - (float)(jw * 7 + 3)) * 7.0f;   // * pW
    float sp = sqrtf(hd * hd + wd * wd);
    float prox = expf(-sp * (1.0f / 32.0f)) * expf(-td * 0.5f);
    sv[j] = sqrtf(dn * dsh[j] + 1e-8f) + prox;
  }
  // local window with g = int(sqrt(32)) = 5 (replicates reference quirk)
  unsigned m = 0u;
#pragma unroll
  for (int dt = -1; dt <= 1; ++dt)
#pragma unroll
    for (int dh = -1; dh <= 1; ++dh)
#pragma unroll
      for (int dw = -1; dw <= 1; ++dw) {
        int j = i + dt * 25 + dh * 5 + dw;
        if (j >= 0 && j < NBLK) m |= 1u << j;
      }
  // top-4 excluding diagonal (first-index tie-break like jax top_k)
  unsigned chosen = 1u << i;
#pragma unroll
  for (int p = 0; p < 4; ++p) {
    float best = -1e38f;
    int bi = 0;
#pragma unroll
    for (int j = 0; j < NBLK; ++j)
      if (!((chosen >> j) & 1u) && sv[j] > best) { best = sv[j]; bi = j; }
    chosen |= 1u << bi;
    m |= 1u << bi;
  }
  rowmask[i] = m;
}

// tokens(3136x256) x w_qkv^T(256->768) -> qkv f16, one wave per 16x16 tile
__global__ void k_qkvgemm(const _Float16* __restrict__ tok, const _Float16* __restrict__ wq,
                          _Float16* __restrict__ qkv) {
  int tile = blockIdx.x;
  int mt = tile % MTILES;
  int nt = tile / MTILES;  // 0..47
  int lane = threadIdx.x;
  v8f acc = {};
#pragma unroll
  for (int kt = 0; kt < CDIM / 32; ++kt) {
    v16h a = load_a16(tok + (size_t)mt * 16 * CDIM + kt * 32, CDIM, lane);
    v16h b = load_bT16(wq + (size_t)nt * 16 * CDIM + kt * 32, CDIM, lane);
    acc = WMMA_F16(a, b, acc);
  }
  int m0 = mt * 16 + ((lane < 16) ? 0 : 8);
  int nn = nt * 16 + (lane & 15);
#pragma unroll
  for (int r = 0; r < 8; ++r)
    qkv[(size_t)(m0 + r) * QKVD + nn] = (_Float16)acc[r];
}

// flash attention: one wave per (head, 16-row query tile)
// V chunk is staged global->LDS via async-LDS loads (ASYNCcnt), issued at loop
// top and consumed after QK^T + softmax so the copy overlaps compute.
__global__ void k_attn(const _Float16* __restrict__ qkv, const unsigned* __restrict__ rowmask,
                       _Float16* __restrict__ obuf) {
  __shared__ _Float16 ldsP[16 * 32];
  __shared__ _Float16 ldsV[32 * 32];
  int head = blockIdx.x / MTILES;
  int qt   = blockIdx.x % MTILES;
  int lane = threadIdx.x;
  int half8 = (lane < 16) ? 0 : 8;
  int col = lane & 15;

  v16h qa = load_a16(qkv + (size_t)qt * 16 * QKVD + head * HD, QKVD, lane);

  unsigned mrow[8];
#pragma unroll
  for (int r = 0; r < 8; ++r) mrow[r] = rowmask[(qt * 16 + r + half8) / VTOK];

  float m_r[8], l_r[8];
  v8f o0 = {}, o1 = {};
#pragma unroll
  for (int r = 0; r < 8; ++r) { m_r[r] = -1e30f; l_r[r] = 0.f; }

  // per-lane LDS byte offset of this lane's V row (row stride = 32 halves = 64B)
  unsigned vlds = (unsigned)(unsigned long long)(&ldsV[lane * 32]);

  for (int kc = 0; kc < LTOK / 32; ++kc) {
    int j0 = kc * 32;
    // async-stage V chunk (32 keys x 32 dims): lane copies row j0+lane (64B)
    {
      unsigned long long vg =
          (unsigned long long)(qkv + (size_t)(j0 + lane) * QKVD + 2 * CDIM + head * HD);
      asm volatile("s_wait_dscnt 0x0" ::: "memory");  // prior iter's LDS reads done
      asm volatile(
          "global_load_async_to_lds_b128 %0, %1, off\n\t"
          "global_load_async_to_lds_b128 %0, %1, off offset:16\n\t"
          "global_load_async_to_lds_b128 %0, %1, off offset:32\n\t"
          "global_load_async_to_lds_b128 %0, %1, off offset:48"
          :: "v"(vlds), "v"(vg) : "memory");
    }
    if (kc + 1 < LTOK / 32)  // prefetch next K chunk
      __builtin_prefetch(qkv + (size_t)(j0 + 32 + lane) * QKVD + CDIM + head * HD, 0, 3);

    // S = Q K^T for two 16-key tiles
    v16h kb0 = load_bT16(qkv + (size_t)j0 * QKVD + CDIM + head * HD, QKVD, lane);
    v16h kb1 = load_bT16(qkv + (size_t)(j0 + 16) * QKVD + CDIM + head * HD, QKVD, lane);
    v8f s0 = {}, s1 = {};
    s0 = WMMA_F16(qa, kb0, s0);
    s1 = WMMA_F16(qa, kb1, s1);

    int bj0 = (j0 + col) / VTOK;
    int bj1 = (j0 + 16 + col) / VTOK;
#pragma unroll
    for (int r = 0; r < 8; ++r) {
      float v0 = ((mrow[r] >> bj0) & 1u) ? s0[r] * SCALE : -1e30f;
      float v1 = ((mrow[r] >> bj1) & 1u) ? s1[r] * SCALE : -1e30f;
      float rmax = fmaxf(v0, v1);
#pragma unroll
      for (int off = 1; off < 16; off <<= 1) rmax = fmaxf(rmax, __shfl_xor(rmax, off, 32));
      float mnew = fmaxf(m_r[r], rmax);
      float corr = __expf(m_r[r] - mnew);
      float p0 = (v0 > -1e29f) ? __expf(v0 - mnew) : 0.f;
      float p1 = (v1 > -1e29f) ? __expf(v1 - mnew) : 0.f;
      float rsum = p0 + p1;
#pragma unroll
      for (int off = 1; off < 16; off <<= 1) rsum += __shfl_xor(rsum, off, 32);
      l_r[r] = l_r[r] * corr + rsum;
      m_r[r] = mnew;
      o0[r] *= corr;
      o1[r] *= corr;
      int m = r + half8;
      ldsP[m * 32 + col]      = (_Float16)p0;
      ldsP[m * 32 + col + 16] = (_Float16)p1;
    }
    // V async copy + P stores must be visible before the PV WMMAs
    asm volatile("s_wait_asynccnt 0x0" ::: "memory");
    __syncthreads();
    // O += P V (P: 16x32 A-frag from LDS; V: 32x32 -> two B-frags)
    v16h pa  = load_a16(ldsP, 32, lane);
    v16h vb0 = load_b16(ldsV, 32, lane);
    v16h vb1 = load_b16(ldsV + 16, 32, lane);
    o0 = WMMA_F16(pa, vb0, o0);
    o1 = WMMA_F16(pa, vb1, o1);
  }
#pragma unroll
  for (int r = 0; r < 8; ++r) {
    float inv = 1.0f / l_r[r];
    size_t row = (size_t)(qt * 16 + r + half8) * CDIM + head * HD;
    obuf[row + col]      = (_Float16)(o0[r] * inv);
    obuf[row + col + 16] = (_Float16)(o1[r] * inv);
  }
}

// O(3136x256) x w_proj^T + b_proj -> d_out with inverse patch rearrange (f32)
__global__ void k_proj(const _Float16* __restrict__ obuf, const _Float16* __restrict__ wp,
                       const float* __restrict__ bias, float* __restrict__ out) {
  int tile = blockIdx.x;
  int mt = tile % MTILES;
  int nt = tile / MTILES;  // 0..15
  int lane = threadIdx.x;
  v8f acc = {};
#pragma unroll
  for (int kt = 0; kt < CDIM / 32; ++kt) {
    v16h a = load_a16(obuf + (size_t)mt * 16 * CDIM + kt * 32, CDIM, lane);
    v16h b = load_bT16(wp + (size_t)nt * 16 * CDIM + kt * 32, CDIM, lane);
    acc = WMMA_F16(a, b, acc);
  }
  int c = nt * 16 + (lane & 15);
  float bv = bias[c];
#pragma unroll
  for (int r = 0; r < 8; ++r) {
    int l = mt * 16 + r + ((lane < 16) ? 0 : 8);
    int n = l / VTOK, v = l % VTOK;
    int t = v / 49, rem = v % 49, hh = rem / 7, ww = rem % 7;
    int ntb = n >> 4, nh = (n >> 2) & 3, nw = n & 3;
    int T = ntb * 2 + t, H = nh * 7 + hh, W = nw * 7 + ww;
    out[((size_t)(T * 28 + H) * 28 + W) * CDIM + c] = acc[r] + bv;
  }
}

// ---- host launch -----------------------------------------------------------

extern "C" void kernel_launch(void* const* d_in, const int* in_sizes, int n_in,
                              void* d_out, int out_size, void* d_ws, size_t ws_size,
                              hipStream_t stream) {
  (void)in_sizes; (void)n_in; (void)out_size; (void)ws_size;
  const float* x      = (const float*)d_in[0];
  const float* w_qkv  = (const float*)d_in[1];
  const float* w_proj = (const float*)d_in[2];
  const float* b_proj = (const float*)d_in[3];
  float* out = (float*)d_out;

  char* ws = (char*)d_ws;
  size_t off = 0;
  auto carve = [&](size_t bytes) -> char* {
    char* p = ws + off;
    off += (bytes + 255) & ~(size_t)255;
    return p;
  };
  _Float16* tok_h  = (_Float16*)carve((size_t)LTOK * CDIM * 2);
  _Float16* qkv_h  = (_Float16*)carve((size_t)LTOK * QKVD * 2);
  _Float16* o_h    = (_Float16*)carve((size_t)LTOK * CDIM * 2);
  _Float16* wq_h   = (_Float16*)carve((size_t)QKVD * CDIM * 2);
  _Float16* wp_h   = (_Float16*)carve((size_t)CDIM * CDIM * 2);
  float*    dens   = (float*)carve(NBLK * sizeof(float));
  unsigned* rmask  = (unsigned*)carve(NBLK * sizeof(unsigned));

  k_zero<<<1, 32, 0, stream>>>(dens);
  {
    int tot = QKVD * CDIM;  // covers both (wproj is smaller)
    k_convw<<<(tot + 255) / 256, 256, 0, stream>>>(w_qkv, w_proj, wq_h, wp_h);
  }
  k_prep<<<LTOK, CDIM, 0, stream>>>(x, tok_h, dens);
  k_mask<<<1, 32, 0, stream>>>(dens, rmask);
  k_qkvgemm<<<MTILES * (QKVD / 16), 32, 0, stream>>>(tok_h, wq_h, qkv_h);
  k_attn<<<HEADS * MTILES, 32, 0, stream>>>(qkv_h, rmask, o_h);
  k_proj<<<MTILES * (CDIM / 16), 32, 0, stream>>>(o_h, wp_h, b_proj, out);
}
